// GCMC_26517128085856
// MI455X (gfx1250) — matrix operand order repaired and checked by reading, not compiled, not run on hardware
//
#include <hip/hip_runtime.h>

// ---------------- problem constants (match reference) ----------------
#define NUM_USERS 100000
#define NUM_ITEMS 50000
#define N_NODES   150000      // = 16 * 9375, tiles evenly
#define DIM       64
#define N_LAYERS  3
#define NNZ       4800000     // divisible by 32
#define OUT_COLS  256         // 64 * (1 + N_LAYERS)
#define NEG_SLOPE 0.01f

typedef __attribute__((ext_vector_type(16))) _Float16 v16h;
typedef __attribute__((ext_vector_type(8)))  float    v8f;

// ---------------------------------------------------------------------
// init: ego = concat(user_emb, item_emb); also write output cols [0,64)
// ---------------------------------------------------------------------
__global__ void gcmc_init(const float* __restrict__ ue,
                          const float* __restrict__ ie,
                          float* __restrict__ ego,
                          float* __restrict__ outp) {
    int idx = blockIdx.x * blockDim.x + threadIdx.x;
    if (idx >= N_NODES * DIM) return;
    int row = idx >> 6;
    int col = idx & 63;
    float v = (row < NUM_USERS) ? ue[idx] : ie[idx - NUM_USERS * DIM];
    ego[idx] = v;
    outp[(size_t)row * OUT_COLS + col] = v;
}

// ---------------------------------------------------------------------
// SpMM: out[row] += val * x[col].
// Each wave owns 32 consecutive edges.  The wave index is forced
// uniform (readfirstlane), so the unrolled descriptor loads are
// uniform-address loads of read-only arrays -> the backend selects
// SMEM s_load (merged into wide s_load_bN), descriptors land in SGPRs.
// No ds_bpermute / readlane broadcasts at all.  All 32 lanes then
// cooperatively gather the 256B source row (2 cachelines, SGPR-based
// global_load_b64) and scatter 64 f32 L2 atomics per edge
// (global_atomic_add_f32; both tables are L2-resident: 38.4MB each
// << 192MB L2).
// ---------------------------------------------------------------------
__global__ void __launch_bounds__(256)
gcmc_spmm(const int*   __restrict__ rows,
          const int*   __restrict__ cols,
          const float* __restrict__ vals,
          const float* __restrict__ x,
          float*       __restrict__ out) {
    const int lane = threadIdx.x & 31;
    const unsigned wid = (unsigned)__builtin_amdgcn_readfirstlane(
        (int)(blockIdx.x * (blockDim.x >> 5) + (threadIdx.x >> 5)));
    const unsigned e0 = wid * 32u;              // wave-uniform base edge
    if (e0 >= NNZ) return;                      // never taken (exact grid)

#pragma unroll
    for (int j = 0; j < 32; ++j) {
        const int   rj = rows[e0 + j];          // uniform -> s_load
        const int   cj = cols[e0 + j];          // uniform -> s_load
        const float vj = vals[e0 + j];          // uniform -> s_load
        const float2 xv = *(const float2*)(x + (size_t)cj * DIM + lane * 2);
        float* o = out + (size_t)rj * DIM + lane * 2;
        unsafeAtomicAdd(o,     vj * xv.x);      // global_atomic_add_f32
        unsafeAtomicAdd(o + 1, vj * xv.y);
    }
}

// ---------------------------------------------------------------------
// WMMA fragment helper: two runs of 8 consecutive f32 -> v16h halves
// (halves 0..7 from p0[0..7], halves 8..15 from p1[0..7])
// ---------------------------------------------------------------------
__device__ __forceinline__ v16h load2x8_f16(const float* __restrict__ p0,
                                            const float* __restrict__ p1) {
    float4 x0 = *(const float4*)(p0);
    float4 x1 = *(const float4*)(p0 + 4);
    float4 y0 = *(const float4*)(p1);
    float4 y1 = *(const float4*)(p1 + 4);
    v16h a;
    a[0]  = (_Float16)x0.x; a[1]  = (_Float16)x0.y;
    a[2]  = (_Float16)x0.z; a[3]  = (_Float16)x0.w;
    a[4]  = (_Float16)x1.x; a[5]  = (_Float16)x1.y;
    a[6]  = (_Float16)x1.z; a[7]  = (_Float16)x1.w;
    a[8]  = (_Float16)y0.x; a[9]  = (_Float16)y0.y;
    a[10] = (_Float16)y0.z; a[11] = (_Float16)y0.w;
    a[12] = (_Float16)y1.x; a[13] = (_Float16)y1.y;
    a[14] = (_Float16)y1.z; a[15] = (_Float16)y1.w;
    return a;
}

// ---------------------------------------------------------------------
// Fused dense stage for one layer, one 16-row tile per block (4 waves):
//   ego  = leaky_relu(side @ Gw^T + gb)       (written to ego buffer)
//   mlp  = ego @ Bw^T + bb                    (written to out cols [co,co+64))
// Each wave computes one 16x16 tile with v_wmma_f32_16x16x32_f16 (K=64
// as two K=32 steps, f32 accumulate).  Intermediate exchanged via LDS.
// ---------------------------------------------------------------------
#define LDS_STRIDE 72   // 64 + 8 halfs of pad

__global__ void __launch_bounds__(128)
gcmc_dense(const float* __restrict__ side,
           float*       __restrict__ ego,
           float*       __restrict__ outp,
           const float* __restrict__ gw,  // 64x64, row n contiguous in k
           const float* __restrict__ gb,  // 64
           const float* __restrict__ bw,  // 64x64
           const float* __restrict__ bb,  // 64
           int col_off) {
    __shared__ _Float16 egoT[16 * LDS_STRIDE];

    const int tid  = threadIdx.x;
    const int wave = tid >> 5;
    const int lane = tid & 31;
    const int h    = lane >> 4;   // half-wave selector in WMMA layouts
    const int m    = lane & 15;   // A: row-in-tile; B/D: col-in-tile
    const int n0   = wave * 16;   // this wave's output column tile
    const int n    = n0 + m;      // output column owned by this lane (B/D)
    const int row0 = blockIdx.x * 16;

    // ---- GEMM 1: acc = side_tile @ gw^T ----
    v8f acc = {0.f, 0.f, 0.f, 0.f, 0.f, 0.f, 0.f, 0.f};
#pragma unroll
    for (int kb = 0; kb < DIM; kb += 32) {
        // A (16x32 f16): lane(m,h): halves 0..7 <- K=kb+8h+i, 8..15 <- K=kb+16+8h+i
        const float* ap = side + (size_t)(row0 + m) * DIM + kb + h * 8;
        v16h afrag = load2x8_f16(ap, ap + 16);
        // B (32x16 f16): lane(n,h): halves 0..7 <- K=kb+8h+i, 8..15 <- K=kb+16+8h+i
        const float* bp = gw + (size_t)n * DIM + kb + h * 8;
        v16h bfrag = load2x8_f16(bp, bp + 16);
        acc = __builtin_amdgcn_wmma_f32_16x16x32_f16(
                  false, afrag, false, bfrag, (short)0, acc, false, false);
    }

    // bias + leaky_relu; D layout: VGPR v -> row m = v + 8h, col n
    const float gbias = gb[n];
#pragma unroll
    for (int v = 0; v < 8; ++v) {
        int   mr = v + 8 * h;
        float x  = acc[v] + gbias;
        x = (x > 0.f) ? x : NEG_SLOPE * x;
        ego[(size_t)(row0 + mr) * DIM + n] = x;          // next layer's SpMM input
        egoT[mr * LDS_STRIDE + n] = (_Float16)x;         // for GEMM 2 A-fragments
    }
    __syncthreads();

    // ---- GEMM 2: acc2 = ego_tile @ bw^T ----
    v8f acc2 = {0.f, 0.f, 0.f, 0.f, 0.f, 0.f, 0.f, 0.f};
#pragma unroll
    for (int kb = 0; kb < DIM; kb += 32) {
        v16h afrag;
        const _Float16* lp = &egoT[m * LDS_STRIDE + kb + h * 8];
#pragma unroll
        for (int i = 0; i < 8; ++i) {
            afrag[i]     = lp[i];        // K = kb + 8h + i
            afrag[8 + i] = lp[16 + i];   // K = kb + 16 + 8h + i
        }
        const float* bp = bw + (size_t)n * DIM + kb + h * 8;
        v16h bfrag = load2x8_f16(bp, bp + 16);
        acc2 = __builtin_amdgcn_wmma_f32_16x16x32_f16(
                   false, afrag, false, bfrag, (short)0, acc2, false, false);
    }

    const float bbias = bb[n];
#pragma unroll
    for (int v = 0; v < 8; ++v) {
        int mr = v + 8 * h;
        outp[(size_t)(row0 + mr) * OUT_COLS + col_off + n] = acc2[v] + bbias;
    }
}

// ---------------------------------------------------------------------
extern "C" void kernel_launch(void* const* d_in, const int* in_sizes, int n_in,
                              void* d_out, int out_size, void* d_ws, size_t ws_size,
                              hipStream_t stream) {
    const float* ue = (const float*)d_in[0];   // (100000,64)
    const float* ie = (const float*)d_in[1];   // (50000,64)
    const float* gw = (const float*)d_in[2];   // (3,64,64)
    const float* gb = (const float*)d_in[3];   // (3,64)
    const float* bw = (const float*)d_in[4];   // (3,64,64)
    const float* bb = (const float*)d_in[5];   // (3,64)
    const int*   er = (const int*)d_in[6];     // (NNZ,)
    const int*   ec = (const int*)d_in[7];     // (NNZ,)
    const float* ev = (const float*)d_in[8];   // (NNZ,)
    float* outp = (float*)d_out;               // (150000,256) row-major

    // workspace: ego (38.4MB) + side (38.4MB)
    float* ego  = (float*)d_ws;
    float* side = ego + (size_t)N_NODES * DIM;

    {
        int total = N_NODES * DIM;
        gcmc_init<<<(total + 255) / 256, 256, 0, stream>>>(ue, ie, ego, outp);
    }

    // one wave (32 lanes) per 32 edges; 8 waves per block
    const int spmm_waves  = NNZ / 32;                 // 150000
    const int spmm_blocks = (spmm_waves + 7) / 8;     // 18750

    for (int l = 0; l < N_LAYERS; ++l) {
        hipMemsetAsync(side, 0, (size_t)N_NODES * DIM * sizeof(float), stream);
        gcmc_spmm<<<spmm_blocks, 256, 0, stream>>>(er, ec, ev, ego, side);
        gcmc_dense<<<N_NODES / 16, 128, 0, stream>>>(
            side, ego, outp,
            gw + (size_t)l * DIM * DIM, gb + (size_t)l * DIM,
            bw + (size_t)l * DIM * DIM, bb + (size_t)l * DIM,
            DIM * (l + 1));
    }
}
